// MYModel_21560735826289
// MI455X (gfx1250) — matrix operand order repaired and checked by reading, not compiled, not run on hardware
//
#include <hip/hip_runtime.h>
#include <hip/hip_bf16.h>

typedef __attribute__((ext_vector_type(16))) _Float16 v16h;
typedef __attribute__((ext_vector_type(8)))  float    v8f;

#define WAVES_PER_BLOCK 8
#define ROWS_PER_WAVE   16
#define ROWS_PER_BLOCK  (WAVES_PER_BLOCK * ROWS_PER_WAVE)   // 128
#define T_STEPS         5
#define UNITS           64
#define NGATES          256   // 4 * UNITS
#define NTILES          16    // NGATES / 16

__device__ __forceinline__ float sigmoid_f(float x) {
    return 1.0f / (1.0f + __expf(-x));
}
__device__ __forceinline__ float tanh_f(float x) {
    // stable tanh via exp of negative magnitude
    float t = __expf(-2.0f * fabsf(x));
    float r = (1.0f - t) / (1.0f + t);
    return copysignf(r, x);
}

// Load a pre-swizzled B fragment (8 dwords per lane) from LDS
__device__ __forceinline__ v16h ld_bfrag(const unsigned int* p) {
    v16h f;
    unsigned int* fp = (unsigned int*)&f;
#pragma unroll
    for (int v = 0; v < 8; ++v) fp[v] = p[v];
    return f;
}

__global__ __launch_bounds__(256) void
lstm_wmma_kernel(const float* __restrict__ x,   // [B, T, 1]
                 const float* __restrict__ W,   // [1, 256]
                 const float* __restrict__ U,   // [64, 256]
                 const float* __restrict__ b,   // [256]
                 const float* __restrict__ Wd,  // [64, 1]
                 const float* __restrict__ bd,  // [1]
                 float* __restrict__ out)       // [B, 1]
{
    // LDS: B-fragments of U (f16, WMMA-swizzled): [ntile][kb][lane][8 dwords]
    __shared__ unsigned int s_Bfrag[NTILES * 2 * 32 * 8];          // 32 KB
    __shared__ _Float16     s_h[WAVES_PER_BLOCK][ROWS_PER_WAVE * UNITS]; // 16 KB
    __shared__ float        s_x[WAVES_PER_BLOCK][ROWS_PER_WAVE * T_STEPS]; // 2.5 KB
    __shared__ float        s_red[WAVES_PER_BLOCK][ROWS_PER_WAVE * 16];   // 8 KB

    const int tid  = threadIdx.x;
    const int lane = tid & 31;
    const int wave = tid >> 5;
    const int lo   = lane & 15;    // N index within tile / M index for A rows
    const int hi   = lane >> 4;    // lane half
    const int rowBase = (blockIdx.x * WAVES_PER_BLOCK + wave) * ROWS_PER_WAVE;

    // ---- Stage U into LDS as packed f16 B fragments (cooperative, whole block)
    // B 32x16 f16 layout: VGPR v, half t: K = 2*(v&3) + 8*hi + 16*(v>>2) + t; N = lane&15
    for (int e = tid; e < NTILES * 2 * 32; e += 256) {
        const int nt  = e >> 6;
        const int kb  = (e >> 5) & 1;
        const int ln  = e & 31;
        const int llo = ln & 15;
        const int lhi = ln >> 4;
        const int col = nt * 16 + llo;
        unsigned int* dst = &s_Bfrag[e * 8];
#pragma unroll
        for (int v = 0; v < 8; ++v) {
            const int r0 = kb * 32 + 2 * (v & 3) + 8 * lhi + 16 * (v >> 2);
            union { _Float16 f[2]; unsigned int u; } p;
            p.f[0] = (_Float16)U[r0 * NGATES + col];
            p.f[1] = (_Float16)U[(r0 + 1) * NGATES + col];
            dst[v] = p.u;
        }
    }
    // ---- Stage this wave's x tile (16 rows x 5 timesteps, contiguous 80 floats)
    for (int i = lane; i < ROWS_PER_WAVE * T_STEPS; i += 32)
        s_x[wave][i] = x[rowBase * T_STEPS + i];

    // ---- Per-lane gate weights/bias: column n = nt*16 + lo
    float wn[NTILES], bn[NTILES];
#pragma unroll
    for (int nt = 0; nt < NTILES; ++nt) {
        wn[nt] = W[nt * 16 + lo];
        bn[nt] = b[nt * 16 + lo];
    }
    float wd[4];
#pragma unroll
    for (int g = 0; g < 4; ++g) wd[g] = Wd[g * 16 + lo];

    __syncthreads();

    // Cell state fragments: 4 unit-groups x 8 rows (C/D layout: m = hi*8 + r, unit = g*16 + lo)
    float c_frag[4][8];

    // ================= t = 0 : h0 = 0, so z = x0*W + b (no WMMA) =================
#pragma unroll
    for (int g = 0; g < 4; ++g) {
#pragma unroll
        for (int r = 0; r < 8; ++r) {
            const int m = hi * 8 + r;
            const float xv = s_x[wave][m * T_STEPS + 0];
            const float zi = fmaf(xv, wn[g],      bn[g]);
            const float zf = fmaf(xv, wn[g + 4],  bn[g + 4]);
            const float zg = fmaf(xv, wn[g + 8],  bn[g + 8]);
            const float zo = fmaf(xv, wn[g + 12], bn[g + 12]);
            const float iv = sigmoid_f(zi);
            const float gv = tanh_f(zg);
            const float ov = sigmoid_f(zo);
            (void)zf; // f * c0 == 0
            const float cv = iv * gv;
            c_frag[g][r] = cv;
            const float hv = ov * tanh_f(cv);
            s_h[wave][m * UNITS + g * 16 + lo] = (_Float16)hv;
        }
    }
    __syncthreads();

    // ================= t = 1 .. 4 : z = x_t*W + h@U + b via WMMA =================
    for (int t = 1; t < T_STEPS; ++t) {
        // Load A fragments (h tile, 16x64 f16 -> two 16x32 fragments) from LDS.
        // A 16x32 f16 layout: M = lane&15; VGPR v holds K = 2*(v&3) + 8*hi + 16*(v>>2) (+1)
        v16h a0, a1;
        unsigned int* a0p = (unsigned int*)&a0;
        unsigned int* a1p = (unsigned int*)&a1;
#pragma unroll
        for (int v = 0; v < 8; ++v) {
            const int k = 2 * (v & 3) + 8 * hi + 16 * (v >> 2);
            a0p[v] = *(const unsigned int*)&s_h[wave][lo * UNITS + k];
            a1p[v] = *(const unsigned int*)&s_h[wave][lo * UNITS + 32 + k];
        }

#pragma unroll
        for (int g = 0; g < 4; ++g) {
            v8f zi, zf, zg, zo;
#pragma unroll
            for (int r = 0; r < 8; ++r) {
                const int m = hi * 8 + r;
                const float xv = s_x[wave][m * T_STEPS + t];
                zi[r] = fmaf(xv, wn[g],      bn[g]);
                zf[r] = fmaf(xv, wn[g + 4],  bn[g + 4]);
                zg[r] = fmaf(xv, wn[g + 8],  bn[g + 8]);
                zo[r] = fmaf(xv, wn[g + 12], bn[g + 12]);
            }
            // 2 K-steps (K=64) per gate; B fragments streamed from pre-swizzled LDS
            const int nti = g, ntf = g + 4, ntg = g + 8, nto = g + 12;
            zi = __builtin_amdgcn_wmma_f32_16x16x32_f16(false, a0, false,
                    ld_bfrag(&s_Bfrag[((nti * 2 + 0) * 32 + lane) * 8]), (short)0, zi, false, false);
            zi = __builtin_amdgcn_wmma_f32_16x16x32_f16(false, a1, false,
                    ld_bfrag(&s_Bfrag[((nti * 2 + 1) * 32 + lane) * 8]), (short)0, zi, false, false);
            zf = __builtin_amdgcn_wmma_f32_16x16x32_f16(false, a0, false,
                    ld_bfrag(&s_Bfrag[((ntf * 2 + 0) * 32 + lane) * 8]), (short)0, zf, false, false);
            zf = __builtin_amdgcn_wmma_f32_16x16x32_f16(false, a1, false,
                    ld_bfrag(&s_Bfrag[((ntf * 2 + 1) * 32 + lane) * 8]), (short)0, zf, false, false);
            zg = __builtin_amdgcn_wmma_f32_16x16x32_f16(false, a0, false,
                    ld_bfrag(&s_Bfrag[((ntg * 2 + 0) * 32 + lane) * 8]), (short)0, zg, false, false);
            zg = __builtin_amdgcn_wmma_f32_16x16x32_f16(false, a1, false,
                    ld_bfrag(&s_Bfrag[((ntg * 2 + 1) * 32 + lane) * 8]), (short)0, zg, false, false);
            zo = __builtin_amdgcn_wmma_f32_16x16x32_f16(false, a0, false,
                    ld_bfrag(&s_Bfrag[((nto * 2 + 0) * 32 + lane) * 8]), (short)0, zo, false, false);
            zo = __builtin_amdgcn_wmma_f32_16x16x32_f16(false, a1, false,
                    ld_bfrag(&s_Bfrag[((nto * 2 + 1) * 32 + lane) * 8]), (short)0, zo, false, false);

            // Gate nonlinearity + state update (elementwise: all gates share C-layout positions)
#pragma unroll
            for (int r = 0; r < 8; ++r) {
                const float iv = sigmoid_f(zi[r]);
                const float fv = sigmoid_f(zf[r]);
                const float gv = tanh_f(zg[r]);
                const float ov = sigmoid_f(zo[r]);
                const float cv = fmaf(fv, c_frag[g][r], iv * gv);
                c_frag[g][r] = cv;
                const float hv = ov * tanh_f(cv);
                const int m = hi * 8 + r;
                s_h[wave][m * UNITS + g * 16 + lo] = (_Float16)hv;
            }
        }
        __syncthreads();
    }

    // ================= Dense(1) + ReLU on h_last =================
    // Each lane produces per-row partials over its unit column (lo + 16g)
#pragma unroll
    for (int r = 0; r < 8; ++r) {
        const int m = hi * 8 + r;
        float s = 0.0f;
#pragma unroll
        for (int g = 0; g < 4; ++g)
            s = fmaf((float)s_h[wave][m * UNITS + g * 16 + lo], wd[g], s);
        s_red[wave][m * 16 + lo] = s;
    }
    __syncthreads();

    if (lane < 16) {
        float s = bd[0];
#pragma unroll
        for (int j = 0; j < 16; ++j) s += s_red[wave][lane * 16 + j];
        out[rowBase + lane] = fmaxf(s, 0.0f);
    }
}

extern "C" void kernel_launch(void* const* d_in, const int* in_sizes, int n_in,
                              void* d_out, int out_size, void* d_ws, size_t ws_size,
                              hipStream_t stream) {
    const float* x  = (const float*)d_in[0];
    const float* W  = (const float*)d_in[1];
    const float* U  = (const float*)d_in[2];
    const float* b  = (const float*)d_in[3];
    const float* Wd = (const float*)d_in[4];
    const float* bd = (const float*)d_in[5];
    float* out = (float*)d_out;

    const int B = in_sizes[0] / T_STEPS;          // 262144
    const int blocks = (B + ROWS_PER_BLOCK - 1) / ROWS_PER_BLOCK; // 2048

    lstm_wmma_kernel<<<blocks, 256, 0, stream>>>(x, W, U, b, Wd, bd, out);
}
